// SubjectSpecificProjection_72739566125853
// MI455X (gfx1250) — matrix operand order repaired and checked by reading, not compiled, not run on hardware
//
#include <hip/hip_runtime.h>

#define BATCH     16384
#define EEG_DIM   256
#define CLIP_DIM  512
#define NSUB      13
#define TM        64                    // rows per tile (4 M-subtiles)
#define MAX_TILES (BATCH / TM + NSUB)   // 269
#define APITCH    264   // bf16 pitch for staged eeg tile  (132 dw % 64 = 4)
#define HPITCH    520   // bf16 pitch for h tile           (260 dw % 64 = 4)
#define PPITCH    132   // f32 pitch for norm partials     (132 % 64 = 4)

typedef __attribute__((ext_vector_type(16))) __bf16 v16bf_t;
typedef __attribute__((ext_vector_type(8)))  __bf16 v8bf_t;
typedef __attribute__((ext_vector_type(8)))  float  v8f_t;

// ---------- A-fragment: 16-bit A 16x32 ISA layout ----------
// lane<16 : elems 0..7 = K[k0..k0+7],    elems 8..15 = K[k0+16..k0+23]
// lane>=16: elems 0..7 = K[k0+8..k0+15], elems 8..15 = K[k0+24..k0+31]
template <typename P>
__device__ inline v16bf_t load_a_frag(P base /* row ptr + k0 */, int lane) {
  const __bf16* p = (const __bf16*)base + ((lane < 16) ? 0 : 8);
  v8bf_t lo = *(const v8bf_t*)(p);
  v8bf_t hi = *(const v8bf_t*)(p + 16);
  v16bf_t r;
#pragma unroll
  for (int i = 0; i < 8; ++i) { r[i] = lo[i]; r[8 + i] = hi[i]; }
  return r;
}

// ---------- B-fragment: 16-bit B 32x16 ISA layout ----------
// lane<16 : elems 0..7 = K[k0..k0+7], 8..15 = K[k0+16..k0+23]; lane>=16: K[k0+8..k0+15], K[k0+24..k0+31]
__device__ inline v16bf_t load_b_frag(const __bf16* base /* col ptr + k0 */, int lane) {
  const __bf16* p = base + ((lane < 16) ? 0 : 8);
  v8bf_t lo = *(const v8bf_t*)(p);
  v8bf_t hi = *(const v8bf_t*)(p + 16);
  v16bf_t r;
#pragma unroll
  for (int i = 0; i < 8; ++i) { r[i] = lo[i]; r[8 + i] = hi[i]; }
  return r;
}

// ================= binning =================
__global__ void k_zero(int* counts, int* cursor) {
  if (threadIdx.x < 16) { counts[threadIdx.x] = 0; cursor[threadIdx.x] = 0; }
}

__global__ void k_hist(const int* __restrict__ sid, int* __restrict__ counts, int n) {
  int i = blockIdx.x * blockDim.x + threadIdx.x;
  if (i < n) atomicAdd(&counts[sid[i]], 1);
}

__global__ void k_scan(const int* __restrict__ counts, int* __restrict__ offsets,
                       int* __restrict__ tileStart) {
  if (threadIdx.x == 0 && blockIdx.x == 0) {
    int off = 0, tt = 0;
    for (int s = 0; s < NSUB; ++s) {
      offsets[s] = off; tileStart[s] = tt;
      off += counts[s]; tt += (counts[s] + TM - 1) / TM;
    }
    offsets[NSUB] = off; tileStart[NSUB] = tt;
  }
}

__global__ void k_scatter(const int* __restrict__ sid, const int* __restrict__ offsets,
                          int* __restrict__ cursor, int* __restrict__ idxSorted, int n) {
  int i = blockIdx.x * blockDim.x + threadIdx.x;
  if (i < n) {
    int s = sid[i];
    int p = atomicAdd(&cursor[s], 1);
    idxSorted[offsets[s] + p] = i;
  }
}

// ================= weight transpose: out[s][n][k] = bf16(in[s][k][n]) =================
__global__ void k_transpose(const float* __restrict__ in, __bf16* __restrict__ out,
                            int rows, int cols) {
  long s = blockIdx.y;
  int idx = blockIdx.x * blockDim.x + threadIdx.x;
  if (idx < rows * cols) {
    int k = idx % rows;
    int n = idx / rows;
    out[s * (long)rows * cols + idx] = (__bf16)in[s * (long)rows * cols + (long)k * cols + n];
  }
}

// ================= fused routed MLP (WMMA bf16, 64-row tiles) =================
__global__ __launch_bounds__(256, 1)
void k_fused_mlp(const int* __restrict__ idxSorted, const int* __restrict__ tileStart,
                 const int* __restrict__ offsets, const int* __restrict__ counts,
                 const float* __restrict__ eeg, const __bf16* __restrict__ w1t,
                 const __bf16* __restrict__ w2t, const float* __restrict__ b1,
                 const float* __restrict__ b2, float* __restrict__ out) {
  __shared__ __bf16 aeeg[TM * APITCH];     // 33.8 KB : staged x tile (bf16)
  __shared__ __bf16 hlds[TM * HPITCH];     // 66.6 KB : hidden tile (bf16)
  __shared__ float  partial[TM * PPITCH];  // 33.8 KB : norm partials
  __shared__ float  part2[TM * 4];
  __shared__ float  inv[TM];
  __shared__ int    rowIdx[TM];
  __shared__ int    meta[2];

  const int tid = threadIdx.x;
  if (tid == 0) {
    int t = blockIdx.x;
    int s = -1, row0 = 0;
    if (t < tileStart[NSUB]) {
      s = 0;
      while (s < NSUB - 1 && t >= tileStart[s + 1]) ++s;
      row0 = offsets[s] + (t - tileStart[s]) * TM;
    }
    meta[0] = s; meta[1] = row0;
  }
  __syncthreads();
  const int s = meta[0];
  if (s < 0) return;                 // uniform per block
  const int row0 = meta[1];
  if (tid < TM) {
    int r = row0 + tid;
    rowIdx[tid] = (r < offsets[s] + counts[s]) ? idxSorted[r] : -1;
  }
  __syncthreads();

  // ---- cooperative stage: gathered eeg rows -> bf16 LDS tile (zero-fill pads) ----
  {
    int r  = tid >> 2;          // 0..63
    int c0 = (tid & 3) * 64;    // 4 chunks of 64 floats per row
    int gg = rowIdx[r];
    __bf16* dstrow = &aeeg[r * APITCH + c0];
    if (gg >= 0) {
      const float4* src = (const float4*)(eeg + (long)gg * EEG_DIM + c0);
#pragma unroll
      for (int i = 0; i < 16; ++i) {
        float4 v = src[i];
        dstrow[i * 4 + 0] = (__bf16)v.x;
        dstrow[i * 4 + 1] = (__bf16)v.y;
        dstrow[i * 4 + 2] = (__bf16)v.z;
        dstrow[i * 4 + 3] = (__bf16)v.w;
      }
    } else {
#pragma unroll
      for (int i = 0; i < 64; ++i) dstrow[i] = (__bf16)0.0f;
    }
  }
  __syncthreads();

  const int lane  = tid & 31;
  const int wave  = tid >> 5;
  const int mrow  = lane & 15;         // A-side row within an M-subtile
  const int ncol  = lane & 15;         // B/C-side column within an N-subtile
  const int n0    = wave * 64;         // this wave covers N in [n0, n0+64)
  const int mbase = (lane >> 4) * 8;   // C/D: vgpr r holds M = mbase + r

  // ---------- GEMM1: h = relu(x @ W1 + b1), K = 256, A from LDS ----------
  v8f_t acc[4][4];
#pragma unroll
  for (int mt = 0; mt < 4; ++mt)
#pragma unroll
    for (int sub = 0; sub < 4; ++sub)
#pragma unroll
      for (int e = 0; e < 8; ++e) acc[mt][sub][e] = 0.0f;

  for (int k0 = 0; k0 < EEG_DIM; k0 += 32) {
    v16bf_t a[4];
#pragma unroll
    for (int mt = 0; mt < 4; ++mt)
      a[mt] = load_a_frag(&aeeg[(mt * 16 + mrow) * APITCH + k0], lane);
#pragma unroll
    for (int sub = 0; sub < 4; ++sub) {
      int n = n0 + sub * 16 + ncol;
      v16bf_t b = load_b_frag(w1t + ((long)(s * CLIP_DIM + n)) * EEG_DIM + k0, lane);
#pragma unroll
      for (int mt = 0; mt < 4; ++mt)
        acc[mt][sub] = __builtin_amdgcn_wmma_f32_16x16x32_bf16(
            false, a[mt], false, b, (short)0, acc[mt][sub], false, false);
    }
  }

#pragma unroll
  for (int sub = 0; sub < 4; ++sub) {
    int n = n0 + sub * 16 + ncol;
    float bias = b1[s * CLIP_DIM + n];
#pragma unroll
    for (int mt = 0; mt < 4; ++mt)
#pragma unroll
      for (int r = 0; r < 8; ++r) {
        float v = acc[mt][sub][r] + bias;
        hlds[(mt * 16 + mbase + r) * HPITCH + n] = (__bf16)(v > 0.0f ? v : 0.0f);
      }
  }
  __syncthreads();

  // ---------- GEMM2: y = h @ W2 + b2, K = 512, A from LDS ----------
  v8f_t acc2[4][4];
#pragma unroll
  for (int mt = 0; mt < 4; ++mt)
#pragma unroll
    for (int sub = 0; sub < 4; ++sub)
#pragma unroll
      for (int e = 0; e < 8; ++e) acc2[mt][sub][e] = 0.0f;

  for (int k0 = 0; k0 < CLIP_DIM; k0 += 32) {
    v16bf_t a[4];
#pragma unroll
    for (int mt = 0; mt < 4; ++mt)
      a[mt] = load_a_frag(&hlds[(mt * 16 + mrow) * HPITCH + k0], lane);
#pragma unroll
    for (int sub = 0; sub < 4; ++sub) {
      int n = n0 + sub * 16 + ncol;
      v16bf_t b = load_b_frag(w2t + ((long)(s * CLIP_DIM + n)) * CLIP_DIM + k0, lane);
#pragma unroll
      for (int mt = 0; mt < 4; ++mt)
        acc2[mt][sub] = __builtin_amdgcn_wmma_f32_16x16x32_bf16(
            false, a[mt], false, b, (short)0, acc2[mt][sub], false, false);
    }
  }

  // ---------- bias add (kept in regs) + deterministic row-norm partials ----------
#pragma unroll
  for (int sub = 0; sub < 4; ++sub) {
    int n = n0 + sub * 16 + ncol;
    float bias = b2[s * CLIP_DIM + n];
#pragma unroll
    for (int mt = 0; mt < 4; ++mt)
#pragma unroll
      for (int r = 0; r < 8; ++r) acc2[mt][sub][r] += bias;
  }
#pragma unroll
  for (int mt = 0; mt < 4; ++mt)
#pragma unroll
    for (int r = 0; r < 8; ++r) {
      float p = 0.0f;
#pragma unroll
      for (int sub = 0; sub < 4; ++sub) { float v = acc2[mt][sub][r]; p += v * v; }
      partial[(mt * 16 + mbase + r) * PPITCH + wave * 16 + ncol] = p;
    }
  __syncthreads();

  {
    int row = tid >> 2, q = tid & 3;       // 4 threads reduce 32 partials each
    const float* pr = &partial[row * PPITCH + q * 32];
    float ss = 0.0f;
#pragma unroll
    for (int i = 0; i < 32; ++i) ss += pr[i];
    part2[row * 4 + q] = ss;
  }
  __syncthreads();
  if (tid < TM) {
    float tot = part2[tid * 4] + part2[tid * 4 + 1] + part2[tid * 4 + 2] + part2[tid * 4 + 3];
    inv[tid] = 1.0f / fmaxf(sqrtf(tot), 1e-12f);
  }
  __syncthreads();

  // ---------- scale accumulators and scatter-store ----------
#pragma unroll
  for (int mt = 0; mt < 4; ++mt)
#pragma unroll
    for (int r = 0; r < 8; ++r) {
      int m  = mt * 16 + mbase + r;
      int gg = rowIdx[m];
      if (gg < 0) continue;
      float iv = inv[m];
#pragma unroll
      for (int sub = 0; sub < 4; ++sub) {
        int n = n0 + sub * 16 + ncol;
        out[(long)gg * CLIP_DIM + n] = acc2[mt][sub][r] * iv;
      }
    }
}

// ================= host launcher =================
extern "C" void kernel_launch(void* const* d_in, const int* in_sizes, int n_in,
                              void* d_out, int out_size, void* d_ws, size_t ws_size,
                              hipStream_t stream) {
  const float* eeg = (const float*)d_in[0];
  const int*   sid = (const int*)d_in[1];
  const float* W1  = (const float*)d_in[2];
  const float* b1  = (const float*)d_in[3];
  const float* W2  = (const float*)d_in[4];
  const float* b2  = (const float*)d_in[5];
  float* out = (float*)d_out;

  unsigned char* w = (unsigned char*)d_ws;
  int* counts    = (int*)w; w += 256;
  int* cursor    = (int*)w; w += 256;
  int* offsets   = (int*)w; w += 256;   // NSUB+1 used
  int* tileStart = (int*)w; w += 256;   // NSUB+1 used
  int* idxSorted = (int*)w; w += (size_t)BATCH * sizeof(int);
  __bf16* w1t    = (__bf16*)w; w += (size_t)NSUB * CLIP_DIM * EEG_DIM * sizeof(__bf16);
  __bf16* w2t    = (__bf16*)w; w += (size_t)NSUB * CLIP_DIM * CLIP_DIM * sizeof(__bf16);
  (void)ws_size; (void)in_sizes; (void)n_in; (void)out_size;

  // binning
  k_zero<<<1, 32, 0, stream>>>(counts, cursor);
  k_hist<<<BATCH / 256, 256, 0, stream>>>(sid, counts, BATCH);
  k_scan<<<1, 1, 0, stream>>>(counts, offsets, tileStart);
  k_scatter<<<BATCH / 256, 256, 0, stream>>>(sid, offsets, cursor, idxSorted, BATCH);

  // bf16 transposed weights (WMMA B-layout: contiguous K per column)
  k_transpose<<<dim3((EEG_DIM * CLIP_DIM) / 256, NSUB), 256, 0, stream>>>(W1, w1t, EEG_DIM, CLIP_DIM);
  k_transpose<<<dim3((CLIP_DIM * CLIP_DIM) / 256, NSUB), 256, 0, stream>>>(W2, w2t, CLIP_DIM, CLIP_DIM);

  // fused routed MLP + normalize
  k_fused_mlp<<<MAX_TILES, 256, 0, stream>>>(idxSorted, tileStart, offsets, counts,
                                             eeg, w1t, w2t, b1, b2, out);
}
